// EGNN_DefPlate_84808424227308
// MI455X (gfx1250) — matrix operand order repaired and checked by reading, not compiled, not run on hardware
//
#include <hip/hip_runtime.h>

// ---------------------------------------------------------------------------
// Types for CDNA5 WMMA (wave32, gfx1250)
// ---------------------------------------------------------------------------
typedef __bf16 bf16;
typedef __attribute__((ext_vector_type(16))) __bf16 v16bf;
typedef __attribute__((ext_vector_type(8)))  __bf16 v8bf;
typedef __attribute__((ext_vector_type(8)))  float  v8f;
typedef __attribute__((ext_vector_type(4)))  int    v4i;

static constexpr int N_NODE = 8192;
static constexpr int KNB    = 32;     // neighbors per node (== wave32!)
static constexpr int HID    = 128;
static constexpr int E2P    = 544;    // 2*257 = 514 padded to 17*32
static constexpr int E2D    = 514;
static constexpr int HMLD   = 160;    // [h(128) | msum(16) | pad(16)]
static constexpr int ACT_NONE = 0;
static constexpr int ACT_SILU = 1;

// ---- gfx1250 async global->LDS copy (ASYNCcnt-tracked), with safe fallback --
#if defined(__has_builtin)
#if __has_builtin(__builtin_amdgcn_global_load_async_to_lds_b128) && \
    __has_builtin(__builtin_amdgcn_s_wait_asynccnt)
#define HAVE_ASYNC_LDS 1
#endif
#endif

__device__ __forceinline__ void lds_copy_b128(void* lds_dst, const void* gsrc) {
#ifdef HAVE_ASYNC_LDS
    __builtin_amdgcn_global_load_async_to_lds_b128(
        (__attribute__((address_space(1))) v4i*)gsrc,
        (__attribute__((address_space(3))) v4i*)lds_dst, 0, 0);
#else
    *(v8bf*)lds_dst = *(const v8bf*)gsrc;
#endif
}
__device__ __forceinline__ void async_lds_wait() {
#ifdef HAVE_ASYNC_LDS
    __builtin_amdgcn_s_wait_asynccnt(0);
#endif
}

__device__ __forceinline__ float siluf(float v) {
    return v * (1.0f / (1.0f + __expf(-v)));
}

__device__ __forceinline__ v8f wmma_bf16(v16bf a, v16bf b, v8f c) {
    // D = A(16x32) * B(32x16) + C, f32 accumulate
    return __builtin_amdgcn_wmma_f32_16x16x32_bf16(false, a, false, b, (short)0, c,
                                                   false, false);
}

// A-fragment loader per ISA 16-bit A 16x32 layout:
// lane 0-15: M=lane, halves = K[grp*8 .. +8) and K[16+grp*8 .. +8) with grp=0
// lane16-31: M=lane-16, same with grp=1.
__device__ __forceinline__ v16bf load_a_frag(const bf16* __restrict__ row, int kc, int grp) {
    v8bf lo = *(const v8bf*)(row + kc + grp * 8);
    v8bf hi = *(const v8bf*)(row + kc + 16 + grp * 8);
    v16bf a;
#pragma unroll
    for (int i = 0; i < 8; ++i) { a[i] = lo[i]; a[i + 8] = hi[i]; }
    return a;
}

// B-fragment: weights stored transposed as Wt[N][K]; lane(n=lane&15) reads
// K[kc+grp*8 .. +8) and K[kc+16+grp*8 .. +8) (same K order as A fragment)
__device__ __forceinline__ v16bf load_b_frag(const bf16* __restrict__ wtrow, int kc, int grp) {
    v8bf lo = *(const v8bf*)(wtrow + kc + grp * 8);
    v8bf hi = *(const v8bf*)(wtrow + kc + 16 + grp * 8);
    v16bf b;
#pragma unroll
    for (int i = 0; i < 8; ++i) { b[i] = lo[i]; b[i + 8] = hi[i]; }
    return b;
}

// ---------------------------------------------------------------------------
// Generic node-level GEMM: out = act(A[M x K] @ W[K x N] + bias (+ resid))
// A bf16 row-major (lda), Wt bf16 pre-transposed [Np x Kp] (rows contiguous).
// One wave per 16x16 output tile. grid = (M/16, Np/16), block = 32.
// ---------------------------------------------------------------------------
__global__ void gemm_wmma(const bf16* __restrict__ A, int lda, int Kp,
                          const bf16* __restrict__ Wt,
                          const float* __restrict__ bias, int Nd,
                          int act,
                          const float* __restrict__ resid, int ldr,
                          float* __restrict__ outF, int ldoF,
                          bf16* __restrict__ outB, int ldoB) {
    const int lane  = threadIdx.x;
    const int grp   = lane >> 4;
    const int l15   = lane & 15;
    const int mtile = blockIdx.x;
    const int ntile = blockIdx.y;

    const bf16* Arow = A + (size_t)(mtile * 16 + l15) * lda;
    const bf16* Brow = Wt + (size_t)(ntile * 16 + l15) * Kp;

    v8f acc = {};
    for (int kc = 0; kc < Kp; kc += 32) {
        v16bf a = load_a_frag(Arow, kc, grp);
        v16bf b = load_b_frag(Brow, kc, grp);
        acc = wmma_bf16(a, b, acc);
    }

    const int n = ntile * 16 + l15;
    float bv = (bias != nullptr && n < Nd) ? bias[n] : 0.f;
#pragma unroll
    for (int r = 0; r < 8; ++r) {
        int m = mtile * 16 + grp * 8 + r;           // C layout: lanes16-31 hold M=8..15
        float v = acc[r] + bv;
        if (resid) v += resid[(size_t)m * ldr + n]; // only used when Np==Nd
        if (act == ACT_SILU) v = siluf(v);
        if (outF) outF[(size_t)m * ldoF + n] = v;
        if (outB) outB[(size_t)m * ldoB + n] = (bf16)v;
    }
}

// ---------------------------------------------------------------------------
// Transpose + pad + f32->bf16 a weight matrix W[Kd x Nd] (row-major slice of
// a [.. x Nsrc] source starting at row `ro`) into Wt[Np x Kp] (zero padded).
// ---------------------------------------------------------------------------
__global__ void prep_wt(const float* __restrict__ W, int Nsrc, int ro,
                        int Kd, int Nd, bf16* __restrict__ Wt, int Kp, int Np,
                        int total) {
    int idx = blockIdx.x * 256 + threadIdx.x;
    if (idx >= total) return;
    int n = idx / Kp;
    int k = idx - n * Kp;
    float v = (k < Kd && n < Nd) ? W[(size_t)(ro + k) * Nsrc + n] : 0.f;
    Wt[idx] = (bf16)v;
}

// ---------------------------------------------------------------------------
// Neighbor build: one wave per row; first K positive off-diagonal columns in
// ascending order (matches cumsum<=K + sort in reference). Early exit -> only
// ~2-4 chunks of the 8192-wide row are ever read.
// ---------------------------------------------------------------------------
__global__ void build_nbr(const float* __restrict__ adj, const float* __restrict__ x,
                          int* __restrict__ jl, float* __restrict__ validf,
                          float* __restrict__ rel, float* __restrict__ dsq) {
    __shared__ int sj[8][KNB];
    __shared__ int sfound[8];
    const int wave = threadIdx.x >> 5;
    const int lane = threadIdx.x & 31;
    const int row  = blockIdx.x * 8 + wave;

    int found = 0;
    for (int base = 0; base < N_NODE; base += 32) {
        int col = base + lane;
        float a = adj[(size_t)row * N_NODE + col];
        bool p = (a > 0.f) && (col != row);
        unsigned m32 = (unsigned)__ballot(p);
        int slot = found + __popc(m32 & ((1u << lane) - 1u));
        if (p && slot < KNB) sj[wave][slot] = col;
        found += __popc(m32);
        if (found >= KNB) break;
    }
    if (found > KNB) found = KNB;
    if (lane == 0) sfound[wave] = found;
    __syncthreads();

    found = sfound[wave];
    int k = lane;
    int j = (k < found) ? sj[wave][k] : 0;
    float vf = (k < found) ? 1.f : 0.f;

    float pix = x[(size_t)row * 16 + 3], piy = x[(size_t)row * 16 + 4], piz = x[(size_t)row * 16 + 5];
    float pjx = x[(size_t)j * 16 + 3],   pjy = x[(size_t)j * 16 + 4],   pjz = x[(size_t)j * 16 + 5];
    float rx = pix - pjx, ry = piy - pjy, rz = piz - pjz;

    size_t e = (size_t)row * KNB + k;
    jl[e] = j;
    validf[e] = vf;
    rel[e * 3 + 0] = rx;
    rel[e * 3 + 1] = ry;
    rel[e * 3 + 2] = rz;
    dsq[e] = rx * rx + ry * ry + rz * rz;
}

// Pack input features concat(x[:, :3], x[:, 6:]) -> bf16 [N x 32] (cols 13..31 = 0)
__global__ void pack_x(const float* __restrict__ x, bf16* __restrict__ x13) {
    int i = blockIdx.x * 256 + threadIdx.x;
    const float* xr = x + (size_t)i * 16;
    bf16* o = x13 + (size_t)i * 32;
    o[0] = (bf16)xr[0]; o[1] = (bf16)xr[1]; o[2] = (bf16)xr[2];
#pragma unroll
    for (int c = 0; c < 10; ++c) o[3 + c] = (bf16)xr[6 + c];
#pragma unroll
    for (int c = 13; c < 32; ++c) o[c] = (bf16)0.f;
}

// Zero hm pad columns 144..159 (node-MLP K-padding)
__global__ void zero_hm_pad(bf16* __restrict__ hm) {
    int idx = blockIdx.x * 256 + threadIdx.x;   // over N*16
    int node = idx >> 4, c = 144 + (idx & 15);
    hm[(size_t)node * HMLD + c] = (bf16)0.f;
}

// ---------------------------------------------------------------------------
// Per-node edge MLP (one layer):
//   m1[k] = silu(ga[i] + gb[j_k] + dsq_k * w0c)           (32 x 544 bf16, LDS)
//   m[k]  = silu(m1 @ W1 + b1) * valid_k                  (32 x 16 via WMMA)
//   hm[i, 128..143] = sum_k m[k]
// Block = 64 threads (2 waves, one 16-edge M-tile each). grid = N.
// Weight tile is streamed into LDS with ASYNCcnt-tracked async copies while
// neighbor rows are prefetched toward the WGP.
// ---------------------------------------------------------------------------
__global__ void edge_mlp(const bf16* __restrict__ ga, const bf16* __restrict__ gb,
                         const float* __restrict__ w0c,
                         const bf16* __restrict__ W1t,   // [16 x 544]
                         const float* __restrict__ b1,   // [16]
                         const int* __restrict__ jl, const float* __restrict__ validf,
                         const float* __restrict__ dsq, bf16* __restrict__ hm) {
    __shared__ __align__(32) bf16 sA[32 * E2P];
    __shared__ __align__(32) bf16 sB[16 * E2P];
    __shared__ float sw0c[E2P];
    __shared__ float sb1[16];
    __shared__ float smsum[2][16];
    __shared__ int   sjj[KNB];
    __shared__ float sdq[KNB];

    const int tid = threadIdx.x;
    const int i = blockIdx.x;

    // async weight tile preload (17 x b128 per thread-slot)
    for (int idx = tid * 8; idx < 16 * E2P; idx += 64 * 8)
        lds_copy_b128(&sB[idx], W1t + idx);

    // stage neighbor ids + dsq, prefetch gathered rows into caches
    if (tid < KNB) {
        int j = jl[(size_t)i * KNB + tid];
        sjj[tid] = j;
        sdq[tid] = dsq[(size_t)i * KNB + tid];
        __builtin_prefetch(gb + (size_t)j * E2P, 0, 2);
    }
    for (int c = tid; c < E2P; c += 64) sw0c[c] = (c < E2D) ? w0c[c] : 0.f;
    if (tid < 16) sb1[tid] = b1[tid];
    async_lds_wait();
    __syncthreads();

    // build A tile: vectorized b128 gather (68 chunks of 8 bf16 per edge row)
    const v8bf* gav = (const v8bf*)(ga + (size_t)i * E2P);
    for (int idx = tid; idx < KNB * (E2P / 8); idx += 64) {
        int k  = idx / (E2P / 8);
        int c8 = idx - k * (E2P / 8);
        const v8bf* gbv = (const v8bf*)(gb + (size_t)sjj[k] * E2P);
        float dq = sdq[k];
        v8bf va = gav[c8];
        v8bf vb = gbv[c8];
        v8bf o;
#pragma unroll
        for (int q = 0; q < 8; ++q) {
            float v = (float)va[q] + (float)vb[q] + dq * sw0c[c8 * 8 + q];
            o[q] = (bf16)siluf(v);   // silu(0)=0 keeps K-padding exact
        }
        *(v8bf*)&sA[k * E2P + c8 * 8] = o;
    }
    __syncthreads();

    const int wv = tid >> 5, lane = tid & 31, grp = lane >> 4, l15 = lane & 15;
    const bf16* Ar = &sA[(wv * 16 + l15) * E2P];
    const bf16* Br = &sB[l15 * E2P];
    v8f acc = {};
    for (int kc = 0; kc < E2P; kc += 32) {
        acc = wmma_bf16(load_a_frag(Ar, kc, grp), load_b_frag(Br, kc, grp), acc);
    }

    float part = 0.f;
#pragma unroll
    for (int r = 0; r < 8; ++r) {
        int e = wv * 16 + grp * 8 + r;
        float v = siluf(acc[r] + sb1[l15]) * validf[(size_t)i * KNB + e];
        part += v;
    }
    part += __shfl_xor(part, 16);
    if (lane < 16) smsum[wv][l15] = part;
    __syncthreads();
    if (tid < 16) hm[(size_t)i * HMLD + 128 + tid] = (bf16)(smsum[0][tid] + smsum[1][tid]);
}

// ---------------------------------------------------------------------------
// Per-node phi_x path + equivariant aggregation:
//   t0 = silu(pa[i] + pb[j] + dsq*pc)        (32 x 128, LDS, bias folded in pa)
//   t1 = silu(t0 @ X0 + bx0)                 (32 x 128 via WMMA)
//   w  = t1 . x1w + bx1 ;  nbr[i] = sum_k rel_k * w_k * valid_k
// ---------------------------------------------------------------------------
__global__ void edge_phi_x(const bf16* __restrict__ pa, const bf16* __restrict__ pb,
                           const float* __restrict__ pc,
                           const bf16* __restrict__ X0t,   // [128 x 128]
                           const float* __restrict__ bx0,
                           const float* __restrict__ x1w, const float* __restrict__ bx1,
                           const int* __restrict__ jl, const float* __restrict__ validf,
                           const float* __restrict__ dsq, const float* __restrict__ rel,
                           float* __restrict__ nbr) {
    __shared__ __align__(32) bf16 sA[32 * HID];
    __shared__ __align__(32) bf16 sB[HID * HID];
    __shared__ float sT[32 * HID];
    __shared__ float spc[HID], sbx0[HID], sx1[HID];
    __shared__ int   sjj[KNB];
    __shared__ float sdq[KNB];

    const int tid = threadIdx.x;
    const int i = blockIdx.x;

    for (int idx = tid * 8; idx < HID * HID; idx += 64 * 8)
        lds_copy_b128(&sB[idx], X0t + idx);

    if (tid < KNB) {
        int j = jl[(size_t)i * KNB + tid];
        sjj[tid] = j;
        sdq[tid] = dsq[(size_t)i * KNB + tid];
        __builtin_prefetch(pb + (size_t)j * HID, 0, 2);
    }
    for (int c = tid; c < HID; c += 64) { spc[c] = pc[c]; sbx0[c] = bx0[c]; sx1[c] = x1w[c]; }
    async_lds_wait();
    __syncthreads();

    const v8bf* pav = (const v8bf*)(pa + (size_t)i * HID);
    for (int idx = tid; idx < KNB * (HID / 8); idx += 64) {
        int k  = idx >> 4;          // HID/8 == 16
        int c8 = idx & 15;
        const v8bf* pbv = (const v8bf*)(pb + (size_t)sjj[k] * HID);
        float dq = sdq[k];
        v8bf va = pav[c8];
        v8bf vb = pbv[c8];
        v8bf o;
#pragma unroll
        for (int q = 0; q < 8; ++q) {
            float v = (float)va[q] + (float)vb[q] + dq * spc[c8 * 8 + q];
            o[q] = (bf16)siluf(v);
        }
        *(v8bf*)&sA[k * HID + c8 * 8] = o;
    }
    __syncthreads();

    const int wv = tid >> 5, lane = tid & 31, grp = lane >> 4, l15 = lane & 15;
    const bf16* Ar = &sA[(wv * 16 + l15) * HID];
    for (int nt = 0; nt < 8; ++nt) {
        const bf16* Br = &sB[(nt * 16 + l15) * HID];
        v8f acc = {};
        for (int kc = 0; kc < HID; kc += 32) {
            acc = wmma_bf16(load_a_frag(Ar, kc, grp), load_b_frag(Br, kc, grp), acc);
        }
        int n = nt * 16 + l15;
#pragma unroll
        for (int r = 0; r < 8; ++r) {
            int e = wv * 16 + grp * 8 + r;
            sT[e * HID + n] = siluf(acc[r] + sbx0[n]);
        }
    }
    __syncthreads();

    if (tid < 32) {
        int e = tid;
        float w = bx1[0];
        for (int c = 0; c < HID; ++c) w += sT[e * HID + c] * sx1[c];
        w *= validf[(size_t)i * KNB + e];
        size_t eg = (size_t)i * KNB + e;
        float cx = rel[eg * 3 + 0] * w;
        float cy = rel[eg * 3 + 1] * w;
        float cz = rel[eg * 3 + 2] * w;
#pragma unroll
        for (int off = 16; off; off >>= 1) {
            cx += __shfl_xor(cx, off);
            cy += __shfl_xor(cy, off);
            cz += __shfl_xor(cz, off);
        }
        if (tid == 0) {
            nbr[(size_t)i * 4 + 0] = cx;
            nbr[(size_t)i * 4 + 1] = cy;
            nbr[(size_t)i * 4 + 2] = cz;
        }
    }
}

// Final heads: v_direct (128->3), stress (128->1), combine with c*nbr.
__global__ void head_out(const bf16* __restrict__ tv2, const bf16* __restrict__ s2,
                         const float* __restrict__ nbr,
                         const float* __restrict__ Wv, const float* __restrict__ bv,
                         const float* __restrict__ Ws, const float* __restrict__ bs,
                         float* __restrict__ out) {
    __shared__ float sWv[384], sWs[128], sbv[3], sbs;
    const int tid = threadIdx.x;
    for (int c = tid; c < 384; c += 256) sWv[c] = Wv[c];
    for (int c = tid; c < 128; c += 256) sWs[c] = Ws[c];
    if (tid < 3) sbv[tid] = bv[tid];
    if (tid == 0) sbs = bs[0];
    __syncthreads();

    int i = blockIdx.x * 256 + tid;
    float a0 = sbv[0], a1 = sbv[1], a2 = sbv[2], ss = sbs;
    for (int c = 0; c < HID; ++c) {
        float tv = (float)tv2[(size_t)i * HID + c];
        a0 += tv * sWv[c * 3 + 0];
        a1 += tv * sWv[c * 3 + 1];
        a2 += tv * sWv[c * 3 + 2];
        ss += (float)s2[(size_t)i * HID + c] * sWs[c];
    }
    const float cf = 1.0f / (float)(N_NODE - 1);
    out[(size_t)i * 4 + 0] = a0 + cf * nbr[(size_t)i * 4 + 0];
    out[(size_t)i * 4 + 1] = a1 + cf * nbr[(size_t)i * 4 + 1];
    out[(size_t)i * 4 + 2] = a2 + cf * nbr[(size_t)i * 4 + 2];
    out[(size_t)i * 4 + 3] = ss;
}

// ---------------------------------------------------------------------------
extern "C" void kernel_launch(void* const* d_in, const int* in_sizes, int n_in,
                              void* d_out, int out_size, void* d_ws, size_t ws_size,
                              hipStream_t stream) {
    // Identify adj / x by unique sizes; everything else = params (tree order).
    const float* adj = nullptr;
    const float* x = nullptr;
    const float* P[64];
    int np = 0;
    for (int i = 0; i < n_in; ++i) {
        if (in_sizes[i] == N_NODE * N_NODE) adj = (const float*)d_in[i];
        else if (in_sizes[i] == N_NODE * 16) x = (const float*)d_in[i];
        else if (np < 64) P[np++] = (const float*)d_in[i];
    }
    float* out = (float*)d_out;

    // Workspace carve-out (all 256B aligned)
    char* ws = (char*)d_ws;
    size_t off = 0;
    auto take = [&](size_t bytes) -> char* {
        char* p = ws + off;
        off += (bytes + 255) & ~(size_t)255;
        return p;
    };
    int*   jl     = (int*)  take((size_t)N_NODE * KNB * 4);
    float* validf = (float*)take((size_t)N_NODE * KNB * 4);
    float* rel    = (float*)take((size_t)N_NODE * KNB * 3 * 4);
    float* dsq    = (float*)take((size_t)N_NODE * KNB * 4);
    float* h      = (float*)take((size_t)N_NODE * HID * 4);
    bf16*  hm     = (bf16*) take((size_t)N_NODE * HMLD * 2);
    bf16*  x13    = (bf16*) take((size_t)N_NODE * 32 * 2);
    bf16*  tA     = (bf16*) take((size_t)N_NODE * HID * 2);
    bf16*  tB     = (bf16*) take((size_t)N_NODE * HID * 2);
    bf16*  tC     = (bf16*) take((size_t)N_NODE * HID * 2);
    bf16*  ga     = (bf16*) take((size_t)N_NODE * E2P * 2);
    bf16*  gb     = (bf16*) take((size_t)N_NODE * E2P * 2);
    bf16*  u      = (bf16*) take((size_t)N_NODE * 256 * 2);
    bf16*  pa     = (bf16*) take((size_t)N_NODE * HID * 2);
    bf16*  pb     = (bf16*) take((size_t)N_NODE * HID * 2);
    float* nbr    = (float*)take((size_t)N_NODE * 4 * 4);
    bf16*  wtb    = (bf16*) take((size_t)1100000 * 2);

    size_t woff = 0;
    auto wtake = [&](size_t e) -> bf16* { bf16* p = wtb + woff; woff += e; return p; };
    bf16* in0t = wtake(128 * 32);
    bf16* in1t = wtake(128 * 128);
    bf16* in2t = wtake(128 * 128);
    bf16 *w0at[4], *w0bt[4], *w1t[4], *n0t[4], *n1t[4];
    for (int l = 0; l < 4; ++l) {
        w0at[l] = wtake((size_t)E2P * 128);
        w0bt[l] = wtake((size_t)E2P * 128);
        w1t[l]  = wtake((size_t)16 * E2P);
        n0t[l]  = wtake((size_t)256 * HMLD);
        n1t[l]  = wtake((size_t)128 * 256);
    }
    bf16* pv0t = wtake(128 * 128);
    bf16* pv1t = wtake(128 * 128);
    bf16* peat = wtake(128 * 128);
    bf16* pebt = wtake(128 * 128);
    bf16* x0t  = wtake(128 * 128);
    bf16* st0t = wtake(128 * 128);
    bf16* st1t = wtake(128 * 128);

    auto prep = [&](const float* W, int Nsrc, int ro, int Kd, int Nd, bf16* Wt, int Kp, int Np) {
        int total = Np * Kp;
        prep_wt<<<(total + 255) / 256, 256, 0, stream>>>(W, Nsrc, ro, Kd, Nd, Wt, Kp, Np, total);
    };
    auto gemm = [&](const bf16* A, int lda, int Kp, const bf16* Wt, const float* bias, int Nd,
                    int act, const float* resid, int ldr,
                    float* oF, int ldoF, bf16* oB, int ldoB, int Np) {
        dim3 g(N_NODE / 16, Np / 16);
        gemm_wmma<<<g, 32, 0, stream>>>(A, lda, Kp, Wt, bias, Nd, act, resid, ldr, oF, ldoF, oB, ldoB);
    };

    // --- weight prep ---
    prep(P[33], 128, 0, 13, 128, in0t, 32, 128);
    prep(P[35], 128, 0, 128, 128, in1t, 128, 128);
    prep(P[37], 128, 0, 128, 128, in2t, 128, 128);
    for (int l = 0; l < 4; ++l) {
        const float* e0w = P[l * 8 + 1];   // [257 x 514]
        prep(e0w, E2D, 0,   128, E2D, w0at[l], 128, E2P);
        prep(e0w, E2D, 128, 128, E2D, w0bt[l], 128, E2P);
        prep(P[l * 8 + 3], 16, 0, E2D, 16, w1t[l], E2P, 16);
        prep(P[l * 8 + 5], 256, 0, 144, 256, n0t[l], HMLD, 256);
        prep(P[l * 8 + 7], 128, 0, 256, 128, n1t[l], 256, 128);
    }
    prep(P[41], 128, 0, 128, 128, pv0t, 128, 128);
    prep(P[43], 128, 0, 128, 128, pv1t, 128, 128);
    prep(P[39], 128, 0,   128, 128, peat, 128, 128);
    prep(P[39], 128, 128, 128, 128, pebt, 128, 128);
    prep(P[47], 128, 0, 128, 128, x0t, 128, 128);
    prep(P[51], 128, 0, 128, 128, st0t, 128, 128);
    prep(P[53], 128, 0, 128, 128, st1t, 128, 128);

    // --- graph + inputs ---
    build_nbr<<<N_NODE / 8, 256, 0, stream>>>(adj, x, jl, validf, rel, dsq);
    pack_x<<<N_NODE / 256, 256, 0, stream>>>(x, x13);
    zero_hm_pad<<<(N_NODE * 16) / 256, 256, 0, stream>>>(hm);

    // --- input MLP: 13 -> 128 -> 128 -> 128 ---
    gemm(x13, 32, 32, in0t, P[32], 128, ACT_SILU, nullptr, 0, nullptr, 0, tA, HID, 128);
    gemm(tA, HID, 128, in1t, P[34], 128, ACT_SILU, nullptr, 0, nullptr, 0, tB, HID, 128);
    gemm(tB, HID, 128, in2t, P[36], 128, ACT_NONE, nullptr, 0, h, HID, hm, HMLD, 128);

    // --- EGNN layers ---
    for (int l = 0; l < 4; ++l) {
        const float* e0w = P[l * 8 + 1];
        gemm(hm, HMLD, 128, w0at[l], P[l * 8 + 0], E2D, ACT_NONE, nullptr, 0, nullptr, 0, ga, E2P, E2P);
        gemm(hm, HMLD, 128, w0bt[l], nullptr,      E2D, ACT_NONE, nullptr, 0, nullptr, 0, gb, E2P, E2P);
        edge_mlp<<<N_NODE, 64, 0, stream>>>(ga, gb, e0w + (size_t)256 * E2D, w1t[l],
                                            P[l * 8 + 2], jl, validf, dsq, hm);
        gemm(hm, HMLD, HMLD, n0t[l], P[l * 8 + 4], 256, ACT_SILU, nullptr, 0, nullptr, 0, u, 256, 256);
        gemm(u, 256, 256, n1t[l], P[l * 8 + 6], 128, ACT_NONE, h, HID, h, HID, hm, HMLD, 128);
    }

    // --- phi_v (h -> tB), stress (h -> tC) ---
    gemm(hm, HMLD, 128, pv0t, P[40], 128, ACT_SILU, nullptr, 0, nullptr, 0, tA, HID, 128);
    gemm(tA, HID, 128, pv1t, P[42], 128, ACT_SILU, nullptr, 0, nullptr, 0, tB, HID, 128);
    gemm(hm, HMLD, 128, st0t, P[50], 128, ACT_SILU, nullptr, 0, nullptr, 0, tA, HID, 128);
    gemm(tA, HID, 128, st1t, P[52], 128, ACT_SILU, nullptr, 0, nullptr, 0, tC, HID, 128);

    // --- phi_x path (bias of phi_e_proj folded into pa) ---
    gemm(hm, HMLD, 128, peat, P[38], 128, ACT_NONE, nullptr, 0, nullptr, 0, pa, HID, 128);
    gemm(hm, HMLD, 128, pebt, nullptr, 128, ACT_NONE, nullptr, 0, nullptr, 0, pb, HID, 128);
    edge_phi_x<<<N_NODE, 64, 0, stream>>>(pa, pb, P[39] + (size_t)256 * 128, x0t,
                                          P[46], P[49], P[48], jl, validf, dsq, rel, nbr);

    // --- heads + combine ---
    head_out<<<N_NODE / 256, 256, 0, stream>>>(tB, tC, nbr, P[45], P[44], P[55], P[54], out);
    (void)ws_size; (void)out_size; (void)np; (void)adj;
}
